// AttDownsample_45414984188030
// MI455X (gfx1250) — matrix operand order repaired and checked by reading, not compiled, not run on hardware
//
#include <hip/hip_runtime.h>
#include <cstdint>

// ---------------------------------------------------------------------------
// AttDownsample for MI455X (gfx1250), wave32.
//   x: [32,768,56,56] f32  ->  y: [32,768,28,28] f32
// One workgroup = (n, ho, 16-wide wo tile). 512 threads = 16 waves.
// LDS tile: X[fcol=r*33+j][c] in f16 (99 used rows, padded to 128 zero rows).
// GEMM1 (scores): S = Q * X^T  via v_wmma_f32_16x16x32_f16 (9 MN-tiles x 24 K)
// GEMM2 (merge):  Y = A * X    via v_wmma_f32_16x16x32_f16 (48 N-tiles x 4 K)
// Output retired through LDS with global_store_async_from_lds_b128 (ASYNCcnt).
// ---------------------------------------------------------------------------

typedef _Float16 v8h  __attribute__((ext_vector_type(8)));
typedef _Float16 v16h __attribute__((ext_vector_type(16)));
typedef float    v8f  __attribute__((ext_vector_type(8)));

#define C_DIM   768
#define LDW     776                       // halfs per X row (768 + 8 pad)
#define KPAD    128                       // padded K rows (99 used)

#define XF_OFF  0
#define XF_BYTES (KPAD * LDW * 2)         // 198656
#define QF_OFF  (XF_OFF + XF_BYTES)       // 198656 : Q[16][LDW] f16
#define AF_OFF  (QF_OFF + 16 * LDW * 2)   // 223488 : A[16][128] f16
#define SF_OFF  (AF_OFF + 16 * 128 * 2)   // 227584 : S[16][100] f32
#define AT_OFF  (SF_OFF + 16 * 100 * 4)   // 233984 : attn[16][12] f32
#define YF_OFF  (AT_OFF + 16 * 12 * 4)    // 234752 : Y[16 waves][256] f32
#define SMEM_BYTES (YF_OFF + 16 * 256 * 4) // 251136 B  (< 320 KB/WGP)

__device__ __forceinline__ v16h cat8(v8h lo, v8h hi) {
  return __builtin_shufflevector(lo, hi, 0,1,2,3,4,5,6,7,8,9,10,11,12,13,14,15);
}

__device__ __forceinline__ v8f wmma16(v16h a, v16h b, v8f c) {
  // (neg_a, A, neg_b, B, c_mod, C, reuse_a, reuse_b)
  return __builtin_amdgcn_wmma_f32_16x16x32_f16(false, a, false, b, (short)0, c,
                                                false, false);
}

__global__ __launch_bounds__(512)
void att_downsample_kernel(const float* __restrict__ x, float* __restrict__ out)
{
  extern __shared__ __align__(16) unsigned char smem[];
  _Float16* Xf   = (_Float16*)(smem + XF_OFF);
  _Float16* Qf   = (_Float16*)(smem + QF_OFF);
  _Float16* Af   = (_Float16*)(smem + AF_OFF);
  float*    Sf   = (float*)   (smem + SF_OFF);
  float*    atnf = (float*)   (smem + AT_OFF);
  float*    Yf   = (float*)   (smem + YF_OFF);

  const int tid  = threadIdx.x;
  const int lane = tid & 31;               // wave32
  const int wave = tid >> 5;               // 16 waves
  const int wt   = blockIdx.x;             // 0..1  (wo tile)
  const int ho   = blockIdx.y;             // 0..27
  const int n    = blockIdx.z;             // 0..31
  const int wo0  = wt * 16;
  const int wbase = 2 * wo0 - 1;

  // ---- 0) zero the X tile (boundary pad + K-pad rows must be 0, not NaN) ----
  {
    unsigned* z = (unsigned*)Xf;
    for (int i = tid; i < XF_BYTES / 4; i += 512) z[i] = 0u;
  }
  __syncthreads();

  // ---- 1) stage input tile: f32 global -> f16 LDS, coalesced along w -------
  for (int it = wave; it < C_DIM * 3; it += 16) {
    const int c = it / 3;
    const int r = it - c * 3;
    const int h = 2 * ho - 1 + r;
    if ((unsigned)h < 56u) {
      const float* __restrict__ xrow = x + (((size_t)n * C_DIM + c) * 56 + h) * 56;
      for (int j = lane; j < 33; j += 32) {
        const int w = wbase + j;
        _Float16 v = (_Float16)0.f;
        if ((unsigned)w < 56u) v = (_Float16)xrow[w];
        Xf[(r * 33 + j) * LDW + c] = v;
      }
    }
  }
  __syncthreads();

  // ---- 2) build Q = 3x3 avgpool (count_include_pad) in f16 -----------------
  for (int e = tid; e < 16 * C_DIM; e += 512) {
    const int p = e / C_DIM;
    const int c = e - p * C_DIM;
    float s = 0.f;
    #pragma unroll
    for (int r = 0; r < 3; ++r) {
      const int b = r * 33 + 2 * p;
      s += (float)Xf[(b + 0) * LDW + c] + (float)Xf[(b + 1) * LDW + c]
         + (float)Xf[(b + 2) * LDW + c];
    }
    Qf[p * LDW + c] = (_Float16)(s * (1.f / 9.f));
  }
  __syncthreads();

  // ---- 3) GEMM1: S(16x99) = Q(16x768) * X^T,  9 tiles on waves 0..8 --------
  if (wave < 9) {
    const int r    = wave / 3;
    const int jt   = wave - r * 3;         // N tile within row r
    const int arow = lane & 15;            // A: M row this lane carries
    const int akb  = (lane >> 4) << 3;     // A: K sub-base 0/8
    const int nloc = lane & 15;            // B: column (N)
    int j = jt * 16 + nloc; if (j > 32) j = 32;   // clamp (no OOB, dup cols unused)
    const int col  = r * 33 + j;
    const int bkh  = (lane >> 4) << 3;     // B: K sub-base 0/8
    v8f acc = {};
    for (int kc = 0; kc < 24; ++kc) {
      const int c0 = kc * 32;
      const _Float16* qp = &Qf[arow * LDW + c0 + akb];
      v16h a = cat8(*(const v8h*)qp, *(const v8h*)(qp + 16));
      const _Float16* bp = &Xf[col * LDW + c0 + bkh];
      v16h b = cat8(*(const v8h*)bp, *(const v8h*)(bp + 16));
      acc = wmma16(a, b, acc);
    }
    if (jt * 16 + nloc <= 32) {            // skip clamped duplicates
      const int fcol = r * 33 + jt * 16 + nloc;
      const int mb = (lane >> 4) * 8;
      #pragma unroll
      for (int v = 0; v < 8; ++v) Sf[(mb + v) * 100 + fcol] = acc[v];
    }
  }
  __syncthreads();

  // ---- 4) softmax over the 9 window scores (scores /= MERGE_K=3) -----------
  if (tid < 16) {
    const int p = tid;
    float s[9];
    #pragma unroll
    for (int r = 0; r < 3; ++r)
      #pragma unroll
      for (int d = 0; d < 3; ++d)
        s[r * 3 + d] = Sf[p * 100 + r * 33 + 2 * p + d] * (1.f / 3.f);
    float mx = s[0];
    #pragma unroll
    for (int i = 1; i < 9; ++i) mx = fmaxf(mx, s[i]);
    float sum = 0.f;
    #pragma unroll
    for (int i = 0; i < 9; ++i) { s[i] = __expf(s[i] - mx); sum += s[i]; }
    const float inv = 1.f / sum;
    #pragma unroll
    for (int i = 0; i < 9; ++i) atnf[p * 12 + i] = s[i] * inv;
  }
  __syncthreads();

  // ---- 5) build banded A(16x128) f16 (zeros outside the 9-wide band) -------
  for (int e = tid; e < 16 * 128; e += 512) {
    const int p = e >> 7;
    const int k = e & 127;
    float v = 0.f;
    if (k < 99) {
      const int r = (k >= 66) ? 2 : (k >= 33 ? 1 : 0);
      const int d = (k - r * 33) - 2 * p;
      if ((unsigned)d < 3u) v = atnf[p * 12 + r * 3 + d];
    }
    Af[p * 128 + k] = (_Float16)v;
  }
  __syncthreads();

  // ---- 6) GEMM2: Y(16x768) = A(16x128) * X(128x768); async store results ---
  {
    const int arow = lane & 15;            // A: M row this lane carries
    const int akb  = (lane >> 4) << 3;     // A: K sub-base 0/8
    const int cl   = lane & 15;            // B/C: local channel (N)
    const int bkb  = (lane >> 4) << 3;     // B: K sub-base 0/8
    const int mb   = (lane >> 4) * 8;      // C: M base (output point base)
    float* yb = Yf + wave * 256;
    const unsigned long long obase =
        (unsigned long long)(uintptr_t)(out + ((size_t)n * C_DIM * 784)
                                            + (size_t)ho * 28 + wo0);
    for (int t = wave; t < 48; t += 16) {
      const int c0 = t << 4;
      v8f acc = {};
      for (int kc = 0; kc < 4; ++kc) {
        const int k0 = kc * 32;
        const _Float16* ap = &Af[arow * 128 + k0 + akb];
        v16h a = cat8(*(const v8h*)ap, *(const v8h*)(ap + 16));
        v16h b;
        #pragma unroll
        for (int i = 0; i < 16; ++i)
          b[i] = Xf[(k0 + bkb + i + (i >= 8 ? 8 : 0)) * LDW + (c0 + cl)];
        acc = wmma16(a, b, acc);
      }
      // retire previous async stores before reusing this wave's Y buffer
      asm volatile("s_wait_asynccnt 0" ::: "memory");
      *(float4*)&yb[cl * 16 + mb]     = make_float4(acc[0], acc[1], acc[2], acc[3]);
      *(float4*)&yb[cl * 16 + mb + 4] = make_float4(acc[4], acc[5], acc[6], acc[7]);
      asm volatile("s_wait_dscnt 0" ::: "memory");
      const unsigned lds0 = (unsigned)(uintptr_t)(yb + cl * 16 + mb);
      const int voff = ((c0 + cl) * 784 + mb) * 4;
      if (wo0 + mb + 3 < 28)
        asm volatile("global_store_async_from_lds_b128 %0, %1, %2"
                     :: "v"(voff), "v"(lds0), "s"(obase) : "memory");
      if (wo0 + mb + 7 < 28)
        asm volatile("global_store_async_from_lds_b128 %0, %1, %2"
                     :: "v"(voff + 16), "v"(lds0 + 16), "s"(obase) : "memory");
    }
    asm volatile("s_wait_asynccnt 0" ::: "memory");
  }
}

extern "C" void kernel_launch(void* const* d_in, const int* in_sizes, int n_in,
                              void* d_out, int out_size, void* d_ws, size_t ws_size,
                              hipStream_t stream) {
  (void)in_sizes; (void)n_in; (void)out_size; (void)d_ws; (void)ws_size;
  const float* x = (const float*)d_in[0];
  float* out = (float*)d_out;
  dim3 grid(2, 28, 32);   // (wo tile, ho, n)
  att_downsample_kernel<<<grid, 512, SMEM_BYTES, stream>>>(x, out);
}
